// Resonator_64433099375161
// MI455X (gfx1250) — 16-bit and 8-bit matrix operand order repaired and checked by reading, compiled, not run on hardware
//
#include <hip/hip_runtime.h>
#include <stdint.h>

typedef __attribute__((ext_vector_type(16))) _Float16 v16h;
typedef __attribute__((ext_vector_type(8)))  float    v8f;
typedef __attribute__((ext_vector_type(16))) int      v16i;

constexpr int B_  = 1024;
constexpr int F_  = 4;
constexpr int V_  = 64;
constexpr int D_  = 2048;
constexpr int IT_ = 100;
constexpr int TB_ = 16;
constexpr int NW_ = 8;
constexpr int DW_ = D_ / NW_;

#define FP8_P1 ((signed char)0x38)
#define FP8_M1 ((signed char)0xB8)
#define FP8_OR 0x38383838u


__global__ void prep_cb_kernel(const float* __restrict__ cb,
                               signed char* __restrict__ cb8,
                               _Float16* __restrict__ cbT) {
  int i = blockIdx.x * 256 + threadIdx.x;
  if (i >= F_ * V_ * D_) return;
  int f = i / (V_ * D_);
  int r = i - f * (V_ * D_);
  int v = r / D_;
  int d = r - v * D_;
  bool pos = (cb[i] >= 0.0f);
  cb8[i] = pos ? FP8_P1 : FP8_M1;
  cbT[((size_t)f * D_ + d) * V_ + v] = (_Float16)(pos ? 1.0f : -1.0f);
}

__global__ void prep_inp_kernel(const float* __restrict__ inp,
                                signed char* __restrict__ inp8) {
  int i = blockIdx.x * 256 + threadIdx.x;
  if (i >= B_ * D_) return;
  inp8[i] = (inp[i] >= 0.0f) ? FP8_P1 : FP8_M1;
}

__device__ __forceinline__ int koffA128(int i, int half) {
  return ((i >> 2) << 5) + (half << 4) + ((i & 3) << 2);
}


__global__ __launch_bounds__(256) void resonator_kernel(
    const float*       __restrict__ initEst,
    const signed char* __restrict__ inp8,
    const signed char* __restrict__ cb8,
    const _Float16*    __restrict__ cbT,
    float* __restrict__ outEst,
    int*   __restrict__ outIdx,
    float* __restrict__ outMax,
    int*   __restrict__ outConv)
{
  __shared__ __align__(16) signed char estS[F_ * TB_ * D_];
  __shared__ float simS[TB_ * V_];

  const int tid  = threadIdx.x;
  const int lane = tid & 31;
  const int wave = tid >> 5;
  const int m    = lane & 15;
  const int half = lane >> 4;
  const int n    = m;
  const int b0   = blockIdx.x * TB_;

  for (int i = tid; i < F_ * TB_ * D_; i += 256) {
    int f  = i / (TB_ * D_);
    int r  = i - f * (TB_ * D_);
    int mm = r / D_;
    int d  = r - mm * D_;
    float x = initEst[((size_t)(b0 + mm) * F_ + f) * D_ + d];
    estS[i] = (x >= 0.0f) ? FP8_P1 : FP8_M1;
  }
  __syncthreads();

  unsigned int q[2][16];

  #pragma unroll 1
  for (int it = 0; it < IT_; ++it) {
    #pragma unroll
    for (int kc = 0; kc < 2; ++kc) {
      const int kb = wave * DW_ + kc * 128;
      #pragma unroll
      for (int i = 0; i < 16; ++i) {
        const int d = kb + koffA128(i, half);
        unsigned int x = *(const unsigned int*)(inp8 + ((size_t)(b0 + m) * D_ + d));
        x ^= *(const unsigned int*)(estS + (0 * TB_ + m) * D_ + d);
        x ^= *(const unsigned int*)(estS + (1 * TB_ + m) * D_ + d);
        x ^= *(const unsigned int*)(estS + (2 * TB_ + m) * D_ + d);
        x ^= *(const unsigned int*)(estS + (3 * TB_ + m) * D_ + d);
        q[kc][i] = x;
      }
    }

    #pragma unroll 1
    for (int f = 0; f < F_; ++f) {
      for (int i = tid; i < TB_ * V_; i += 256) simS[i] = 0.0f;
      __syncthreads();

      v8f acc[4] = {};
      #pragma unroll
      for (int kc = 0; kc < 2; ++kc) {
        const int kb = wave * DW_ + kc * 128;
        v16i a;
        #pragma unroll
        for (int i = 0; i < 16; ++i) {
          const int d = kb + koffA128(i, half);
          unsigned int e = *(const unsigned int*)(estS + ((f * TB_ + m) * D_ + d));
          a[i] = (int)((q[kc][i] ^ e) | FP8_OR);
        }
        #pragma unroll
        for (int t = 0; t < 4; ++t) {
          const int4* bp =
              (const int4*)(cb8 + ((size_t)(f * V_ + t * 16 + n) * D_ + kb + half * 16));
          int4 b0v = bp[0];
          int4 b1v = bp[2];
          int4 b2v = bp[4];
          int4 b3v = bp[6];
          v16i b = {b0v.x, b0v.y, b0v.z, b0v.w, b1v.x, b1v.y, b1v.z, b1v.w,
                    b2v.x, b2v.y, b2v.z, b2v.w, b3v.x, b3v.y, b3v.z, b3v.w};
          acc[t] = __builtin_amdgcn_wmma_f32_16x16x128_fp8_fp8(a, b, (short)0, acc[t],
                                                               false, false);
        }
      }
      #pragma unroll
      for (int t = 0; t < 4; ++t)
        #pragma unroll
        for (int r = 0; r < 8; ++r)
          atomicAdd(&simS[(r + 8 * half) * V_ + t * 16 + n], acc[t][r]);
      __syncthreads();

      if (it == IT_ - 1 && tid < TB_) {
        float mx = simS[tid * V_];
        for (int v = 1; v < V_; ++v) {
          float s = simS[tid * V_ + v];
          mx = (s > mx) ? s : mx;
        }
        outMax[(size_t)(b0 + tid) * F_ + f] = mx;
      }

      v16h a2[2];
      #pragma unroll
      for (int ks = 0; ks < 2; ++ks)
        #pragma unroll
        for (int j = 0; j < 16; ++j) {
          const int v = ks * 32 + (half << 4) + j;
          a2[ks][j] = (_Float16)simS[m * V_ + v];
        }
      #pragma unroll 1
      for (int t = 0; t < 16; ++t) {
        const int dcol = wave * DW_ + t * 16 + n;
        v8f c = {};
        #pragma unroll
        for (int ks = 0; ks < 2; ++ks) {
          const v16h* bp =
              (const v16h*)(cbT + ((size_t)(f * D_ + dcol) * V_ + ks * 32 + half * 16));
          v16h bb = *bp;
          c = __builtin_amdgcn_wmma_f32_16x16x32_f16(false, a2[ks], false, bb,
                                                     (short)0, c, false, false);
        }
        #pragma unroll
        for (int r = 0; r < 8; ++r)
          estS[(f * TB_ + r + 8 * half) * D_ + dcol] =
              (c[r] >= 0.0f) ? FP8_P1 : FP8_M1;
      }
      __syncthreads();
    }
  }

  #pragma unroll 1
  for (int f = 0; f < F_; ++f) {
    for (int i = tid; i < TB_ * V_; i += 256) simS[i] = 0.0f;
    __syncthreads();
    v8f acc[4] = {};
    #pragma unroll
    for (int kc = 0; kc < 2; ++kc) {
      const int kb = wave * DW_ + kc * 128;
      v16i a;
      #pragma unroll
      for (int i = 0; i < 16; ++i) {
        const int d = kb + koffA128(i, half);
        a[i] = (int)*(const unsigned int*)(estS + ((f * TB_ + m) * D_ + d));
      }
      #pragma unroll
      for (int t = 0; t < 4; ++t) {
        const int4* bp =
            (const int4*)(cb8 + ((size_t)(f * V_ + t * 16 + n) * D_ + kb + half * 16));
        int4 b0v = bp[0];
        int4 b1v = bp[2];
        int4 b2v = bp[4];
        int4 b3v = bp[6];
        v16i b = {b0v.x, b0v.y, b0v.z, b0v.w, b1v.x, b1v.y, b1v.z, b1v.w,
                  b2v.x, b2v.y, b2v.z, b2v.w, b3v.x, b3v.y, b3v.z, b3v.w};
        acc[t] = __builtin_amdgcn_wmma_f32_16x16x128_fp8_fp8(a, b, (short)0, acc[t],
                                                             false, false);
      }
    }
    #pragma unroll
    for (int t = 0; t < 4; ++t)
      #pragma unroll
      for (int r = 0; r < 8; ++r)
        atomicAdd(&simS[(r + 8 * half) * V_ + t * 16 + n], acc[t][r]);
    __syncthreads();

    if (tid < TB_) {
      float s0 = simS[tid * V_];
      float best = (s0 < 0.0f) ? -s0 : s0;
      int bi = 0;
      for (int v = 1; v < V_; ++v) {
        float s = simS[tid * V_ + v];
        float av = (s < 0.0f) ? -s : s;
        if (av > best) { best = av; bi = v; }
      }
      outIdx[(size_t)(b0 + tid) * F_ + f] = bi;
    }
    __syncthreads();
  }

  for (int i = tid; i < F_ * TB_ * D_; i += 256) {
    int f  = i / (TB_ * D_);
    int r  = i - f * (TB_ * D_);
    int mm = r / D_;
    int d  = r - mm * D_;
    outEst[((size_t)(b0 + mm) * F_ + f) * D_ + d] =
        ((unsigned char)estS[i] & 0x80u) ? -1.0f : 1.0f;
  }
  if (blockIdx.x == 0 && tid == 0) outConv[0] = IT_ - 1;
}


extern "C" void kernel_launch(void* const* d_in, const int* in_sizes, int n_in,
                              void* d_out, int out_size, void* d_ws, size_t ws_size,
                              hipStream_t stream) {
  const float* inp  = (const float*)d_in[0];
  const float* init = (const float*)d_in[1];
  const float* cb   = (const float*)d_in[2];

  signed char* cb8  = (signed char*)d_ws;
  _Float16*    cbT  = (_Float16*)((char*)d_ws + (size_t)F_ * V_ * D_);
  signed char* inp8 = (signed char*)((char*)d_ws + (size_t)F_ * V_ * D_ * 3);

  float* outEst  = (float*)d_out;
  int*   outIdx  = (int*)(outEst + (size_t)B_ * F_ * D_);
  float* outMax  = (float*)(outIdx + B_ * F_);
  int*   outConv = (int*)(outMax + B_ * F_);

  prep_cb_kernel<<<(F_ * V_ * D_ + 255) / 256, 256, 0, stream>>>(cb, cb8, cbT);
  prep_inp_kernel<<<(B_ * D_ + 255) / 256, 256, 0, stream>>>(inp, inp8);
  resonator_kernel<<<B_ / TB_, 256, 0, stream>>>(init, inp8, cb8, cbT,
                                                 outEst, outIdx, outMax, outConv);
}
